// TransformerBlock_32109175505160
// MI455X (gfx1250) — matrix operand order repaired and checked by reading, not compiled, not run on hardware
//
#include <hip/hip_runtime.h>

// ---------------- sizes (fixed by the reference) ----------------
#define BB   2
#define SS   2048
#define DD   1024
#define HH   16
#define DHH  64
#define DFFF 4096
#define BS_  (BB * SS)   // 4096 token rows

// ---------------- types ----------------
typedef __attribute__((ext_vector_type(16))) __bf16 v16bf;
typedef __attribute__((ext_vector_type(8)))  __bf16 v8bf;
typedef __attribute__((ext_vector_type(8)))  float  v8f;
typedef __attribute__((ext_vector_type(4)))  unsigned int u32x4;
typedef __attribute__((ext_vector_type(8)))  int          i32x8;
typedef __attribute__((ext_vector_type(4)))  int          i32x4;

#if defined(__AMDGCN__) && __has_builtin(__builtin_amdgcn_tensor_load_to_lds) && \
    __has_builtin(__builtin_amdgcn_s_wait_tensorcnt)
#define GEMM_USE_TDM 1
#else
#define GEMM_USE_TDM 0
#endif

static __device__ __forceinline__ unsigned short f2bf(float f) {
    unsigned int u = __float_as_uint(f);
    u += 0x7FFFu + ((u >> 16) & 1u);   // round-to-nearest-even
    return (unsigned short)(u >> 16);
}

static __device__ __forceinline__ float fast_rcp(float x) {
#if defined(__AMDGCN__) && __has_builtin(__builtin_amdgcn_rcpf)
    return __builtin_amdgcn_rcpf(x);   // v_rcp_f32, no div-fixup blob
#else
    return 1.0f / x;
#endif
}

static __device__ __forceinline__ v16bf join16(v8bf lo, v8bf hi) {
    v16bf r;
#pragma unroll
    for (int i = 0; i < 8; ++i) { r[i] = lo[i]; r[i + 8] = hi[i]; }
    return r;
}

static __device__ __forceinline__ v16bf ld_frag(const unsigned short* p0,
                                                const unsigned short* p1) {
    return join16(*reinterpret_cast<const v8bf*>(p0),
                  *reinterpret_cast<const v8bf*>(p1));
}

// gelu_tanh(x) == x * sigmoid(1.5957691216 * (x + 0.044715 x^3)) ; branch-free,
// one v_exp_f32 + one v_rcp_f32.
static __device__ __forceinline__ float gelu_fast(float x) {
    float u = x + 0.044715f * x * x * x;
    return x * fast_rcp(1.0f + __expf(-1.5957691216f * u));
}

#define NEG_INF (-__builtin_inff())

#if GEMM_USE_TDM
// TDM: load a [128 rows x 32 bf16] tile from a row-major [rows x K] bf16 tensor
// into LDS with 80-byte row stride (64B data + 16B pad), matching the fragment
// layout. pad_interval code 3 = pad every 64B; pad_amount code 3 = 16B pad.
static __device__ __forceinline__ void tdm_load_tile_2d(
    unsigned lds_off, unsigned long long gaddr, int K, int rows) {
    u32x4 g0;
    g0[0] = 1u;                                         // count=1, user mode
    g0[1] = lds_off;                                    // lds_addr
    g0[2] = (unsigned)(gaddr & 0xffffffffull);          // global_addr[31:0]
    g0[3] = (unsigned)((gaddr >> 32) & 0x01ffffffull)   // global_addr[56:32]
          | (2u << 30);                                 // type = 2 ("image")
    i32x8 g1;
    g1[0] = (int)((1u << 16)    // data_size = 2 bytes
                | (1u << 20)    // pad_enable
                | (3u << 22)    // pad_interval: every 16 DWORDs (64B)
                | (3u << 25));  // pad_amount: 4 DWORDs (16B)
    g1[1] = (int)(((unsigned)K & 0xffffu) << 16);       // tensor_dim0 lo16 @ [31:16]
    g1[2] = (int)((((unsigned)K >> 16) & 0xffffu)       // tensor_dim0 hi16
                | (((unsigned)rows & 0xffffu) << 16));  // tensor_dim1 lo16
    g1[3] = (int)(((((unsigned)rows) >> 16) & 0xffffu)  // tensor_dim1 hi16
                | (32u << 16));                         // tile_dim0 = 32 elems
    g1[4] = 128;                                        // tile_dim1 = 128 rows
    g1[5] = K;                                          // tensor_dim0_stride lo32
    g1[6] = 0;
    g1[7] = 0;
    i32x4 z4 = {0, 0, 0, 0};
#if __clang_major__ >= 23
    i32x8 z8 = {0, 0, 0, 0, 0, 0, 0, 0};
    __builtin_amdgcn_tensor_load_to_lds(g0, g1, z4, z4, z8, 0);
#else
    __builtin_amdgcn_tensor_load_to_lds(g0, g1, z4, z4, 0);
#endif
}
#endif

// ---------------- cast kernels ----------------
__global__ __launch_bounds__(256) void cast_bf16_kernel(
    const float* __restrict__ in, unsigned short* __restrict__ out, int n) {
    int i = blockIdx.x * 256 + threadIdx.x;
    if (i < n) out[i] = f2bf(in[i]);
}

// in: [R,C] fp32 row-major -> out: [C,R] bf16 (transposed), LDS-tiled 32x32
__global__ __launch_bounds__(256) void cast_transpose_kernel(
    const float* __restrict__ in, unsigned short* __restrict__ out, int R, int C) {
    __shared__ float tile[32][33];
    const int t = threadIdx.x;
    const int tr0 = blockIdx.y * 32, tc0 = blockIdx.x * 32;
    const int r = t >> 3, c4 = (t & 7) * 4;
    float4 v = *reinterpret_cast<const float4*>(&in[(size_t)(tr0 + r) * C + tc0 + c4]);
    tile[r][c4 + 0] = v.x; tile[r][c4 + 1] = v.y;
    tile[r][c4 + 2] = v.z; tile[r][c4 + 3] = v.w;
    __syncthreads();
    const int cc = r, rr4 = c4;
    unsigned int p0 = (unsigned int)f2bf(tile[rr4 + 0][cc]) |
                      ((unsigned int)f2bf(tile[rr4 + 1][cc]) << 16);
    unsigned int p1 = (unsigned int)f2bf(tile[rr4 + 2][cc]) |
                      ((unsigned int)f2bf(tile[rr4 + 3][cc]) << 16);
    uint2 pk; pk.x = p0; pk.y = p1;
    *reinterpret_cast<uint2*>(&out[(size_t)(tc0 + cc) * R + tr0 + rr4]) = pk;
}

// ---------------- WMMA GEMM: C[M,N] = A[M,K](bf16) * Bt[N,K](bf16)^T ----------------
// double-buffered LDS (TDM DMA when available), 128x128 tile, BK=32, 8 waves
__global__ __launch_bounds__(256) void gemm_bf16_kernel(
    const unsigned short* __restrict__ A,     // [M,K]
    const unsigned short* __restrict__ Bt,    // [N,K] (transposed B)
    const float*          __restrict__ bias,  // [N]
    const float*          __restrict__ resid, // [M,N] or nullptr
    float*                __restrict__ outF,  // [M,N] or nullptr
    unsigned short*       __restrict__ outB,  // [M,N] or nullptr
    int M, int N, int K, int doGelu) {
    __shared__ alignas(16) unsigned short As[2][128 * 40];
    __shared__ alignas(16) unsigned short Bs[2][128 * 40];

    const int tid  = threadIdx.x;
    const int lane = tid & 31, wid = tid >> 5;
    const int half = lane >> 4, l16 = lane & 15;
    const int rowBase = blockIdx.y * 128;
    const int colBase = blockIdx.x * 128;
    const int wm = (wid & 3) * 32;
    const int wn = (wid >> 2) * 64;

    v8f acc[2][4];
#pragma unroll
    for (int i = 0; i < 2; ++i)
#pragma unroll
        for (int j = 0; j < 4; ++j)
#pragma unroll
            for (int r = 0; r < 8; ++r) acc[i][j][r] = 0.0f;

#if GEMM_USE_TDM
    auto issueTiles = [&](int kk, int bi) {
        if (wid == 0) {
            tdm_load_tile_2d((unsigned)(size_t)(void*)&As[bi][0],
                             (unsigned long long)(size_t)A + ((size_t)rowBase * K + kk) * 2ull,
                             K, M);
            tdm_load_tile_2d((unsigned)(size_t)(void*)&Bs[bi][0],
                             (unsigned long long)(size_t)Bt + ((size_t)colBase * K + kk) * 2ull,
                             K, N);
        }
    };
    issueTiles(0, 0);
    if (wid == 0) __builtin_amdgcn_s_wait_tensorcnt(0);
    __syncthreads();
#else
    const int ldRow = tid >> 1;
    const int ldSeg = (tid & 1) * 16;
    const unsigned short* gA = A  + (size_t)(rowBase + ldRow) * K + ldSeg;
    const unsigned short* gB = Bt + (size_t)(colBase + ldRow) * K + ldSeg;
    auto loadSlab = [&](int kk, int bi) {
        *reinterpret_cast<v8bf*>(&As[bi][ldRow * 40 + ldSeg])     = *reinterpret_cast<const v8bf*>(gA + kk);
        *reinterpret_cast<v8bf*>(&As[bi][ldRow * 40 + ldSeg + 8]) = *reinterpret_cast<const v8bf*>(gA + kk + 8);
        *reinterpret_cast<v8bf*>(&Bs[bi][ldRow * 40 + ldSeg])     = *reinterpret_cast<const v8bf*>(gB + kk);
        *reinterpret_cast<v8bf*>(&Bs[bi][ldRow * 40 + ldSeg + 8]) = *reinterpret_cast<const v8bf*>(gB + kk + 8);
    };
    loadSlab(0, 0);
    __syncthreads();
#endif

    int buf = 0;
    for (int kk = 0; kk < K; kk += 32) {
#if GEMM_USE_TDM
        if (kk + 32 < K) issueTiles(kk + 32, buf ^ 1);
#else
        if (kk + 32 < K) loadSlab(kk + 32, buf ^ 1);
        if (kk + 64 < K) {
            __builtin_prefetch(gA + kk + 64, 0, 1);
            __builtin_prefetch(gB + kk + 64, 0, 1);
        }
#endif

        v16bf afrag[2], bfrag[4];
#pragma unroll
        for (int i = 0; i < 2; ++i) {
            int row = wm + i * 16 + l16;
            afrag[i] = ld_frag(&As[buf][row * 40 + half * 8],
                               &As[buf][row * 40 + 16 + half * 8]);
        }
#pragma unroll
        for (int j = 0; j < 4; ++j) {
            int col = wn + j * 16 + l16;
            bfrag[j] = ld_frag(&Bs[buf][col * 40 + half * 8],
                               &Bs[buf][col * 40 + 16 + half * 8]);
        }
#pragma unroll
        for (int i = 0; i < 2; ++i)
#pragma unroll
            for (int j = 0; j < 4; ++j)
                acc[i][j] = __builtin_amdgcn_wmma_f32_16x16x32_bf16(
                    false, afrag[i], false, bfrag[j], (short)0, acc[i][j], false, false);

#if GEMM_USE_TDM
        if (wid == 0) __builtin_amdgcn_s_wait_tensorcnt(0);
#endif
        __syncthreads();
        buf ^= 1;
    }

    // epilogue: +bias, optional gelu, optional residual, fp32/bf16 stores
#pragma unroll
    for (int i = 0; i < 2; ++i) {
#pragma unroll
        for (int j = 0; j < 4; ++j) {
            int mg = rowBase + wm + i * 16 + half * 8;
            int ng = colBase + wn + j * 16 + l16;
            float bv = bias[ng];
#pragma unroll
            for (int r = 0; r < 8; ++r) {
                float v = acc[i][j][r] + bv;
                if (doGelu) v = gelu_fast(v);
                size_t o = (size_t)(mg + r) * N + ng;
                if (resid) v += resid[o];
                if (outF) outF[o] = v;
                if (outB) outB[o] = f2bf(v);
            }
        }
    }
}

// ---------------- flash attention (bf16 WMMA, online softmax, double-buffered K/V) ----------------
__global__ __launch_bounds__(256) void attention_kernel(
    const unsigned short* __restrict__ Q,   // [B*S, D]
    const unsigned short* __restrict__ Km,  // [B*S, D]
    const unsigned short* __restrict__ Vm,  // [B*S, D]
    const unsigned char*  __restrict__ mask,// [B, S] (bool)
    unsigned short*       __restrict__ ctx) // [B*S, D]
{
    __shared__ alignas(16) unsigned short Ks[2][32 * 72];
    __shared__ alignas(16) unsigned short Vt[2][64 * 40];
    __shared__ alignas(16) unsigned short Ps[8 * 16 * 40];

    const int b = blockIdx.z, h = blockIdx.y;
    const int tid  = threadIdx.x;
    const int lane = tid & 31, wid = tid >> 5;
    const int half = lane >> 4, l16 = lane & 15;
    const int q0 = blockIdx.x * 128 + wid * 16;

    const unsigned short* qrow = Q + (size_t)(b * SS + q0 + l16) * DD + h * 64;
    v16bf qf[2];
#pragma unroll
    for (int ko = 0; ko < 2; ++ko)
        qf[ko] = ld_frag(qrow + ko * 32 + half * 8, qrow + ko * 32 + 16 + half * 8);

    float rm[8], rl[8];
    v8f accO[4];
#pragma unroll
    for (int r = 0; r < 8; ++r) { rm[r] = NEG_INF; rl[r] = 0.0f; }
#pragma unroll
    for (int j = 0; j < 4; ++j)
#pragma unroll
        for (int r = 0; r < 8; ++r) accO[j][r] = 0.0f;

    const int lkey = tid >> 3;
    const int lseg = tid & 7;

    auto loadKV = [&](int kb, int bi) {
        size_t rowoff = (size_t)(b * SS + kb + lkey) * DD + h * 64 + lseg * 8;
        *reinterpret_cast<v8bf*>(&Ks[bi][lkey * 72 + lseg * 8]) =
            *reinterpret_cast<const v8bf*>(&Km[rowoff]);
        uint4 raw = *reinterpret_cast<const uint4*>(&Vm[rowoff]);
        unsigned int w[4] = {raw.x, raw.y, raw.z, raw.w};
#pragma unroll
        for (int c = 0; c < 4; ++c) {
            Vt[bi][(lseg * 8 + 2 * c)     * 40 + lkey] = (unsigned short)(w[c] & 0xffffu);
            Vt[bi][(lseg * 8 + 2 * c + 1) * 40 + lkey] = (unsigned short)(w[c] >> 16);
        }
    };

    loadKV(0, 0);
    __syncthreads();

    int buf = 0;
    for (int kb = 0; kb < SS; kb += 32) {
        if (kb + 32 < SS) loadKV(kb + 32, buf ^ 1);

        // --- scores: S[16 x 32] = Q[16 x 64] * K^T[64 x 32], scaled ---
        v8f sc[2];
#pragma unroll
        for (int j = 0; j < 2; ++j) {
            int key = j * 16 + l16;
            v16bf bk0 = ld_frag(&Ks[buf][key * 72 + half * 8],
                                &Ks[buf][key * 72 + 16 + half * 8]);
            v16bf bk1 = ld_frag(&Ks[buf][key * 72 + 32 + half * 8],
                                &Ks[buf][key * 72 + 32 + 16 + half * 8]);
            v8f s;
#pragma unroll
            for (int r = 0; r < 8; ++r) s[r] = 0.0f;
            s = __builtin_amdgcn_wmma_f32_16x16x32_bf16(false, qf[0], false, bk0, (short)0, s, false, false);
            s = __builtin_amdgcn_wmma_f32_16x16x32_bf16(false, qf[1], false, bk1, (short)0, s, false, false);
            float madd = mask[(size_t)b * SS + kb + key] ? NEG_INF : 0.0f;
#pragma unroll
            for (int r = 0; r < 8; ++r) s[r] = s[r] * 0.125f + madd;   // 1/sqrt(64)
            sc[j] = s;
        }

        // --- online softmax update (row = r + 8*half, reduce across 16 lanes) ---
#pragma unroll
        for (int r = 0; r < 8; ++r) {
            float mx = fmaxf(sc[0][r], sc[1][r]);
#pragma unroll
            for (int off = 8; off >= 1; off >>= 1)
                mx = fmaxf(mx, __shfl_xor(mx, off, 32));
            float nm = fmaxf(rm[r], mx);
            float f  = (rm[r] == NEG_INF) ? 0.0f : __expf(rm[r] - nm);
            rl[r] *= f;
#pragma unroll
            for (int j = 0; j < 4; ++j) accO[j][r] *= f;
            rm[r] = nm;
            float p0 = (nm == NEG_INF) ? 0.0f : __expf(sc[0][r] - nm);
            float p1 = (nm == NEG_INF) ? 0.0f : __expf(sc[1][r] - nm);
            float ls = p0 + p1;
#pragma unroll
            for (int off = 8; off >= 1; off >>= 1)
                ls += __shfl_xor(ls, off, 32);
            rl[r] += ls;
            int prow = r + 8 * half;
            Ps[wid * 640 + prow * 40 + l16]      = f2bf(p0);
            Ps[wid * 640 + prow * 40 + 16 + l16] = f2bf(p1);
        }

        // --- O += P[16 x 32] * V[32 x 64] ---
        v16bf pa = ld_frag(&Ps[wid * 640 + l16 * 40 + half * 8],
                           &Ps[wid * 640 + l16 * 40 + 16 + half * 8]);
#pragma unroll
        for (int jn = 0; jn < 4; ++jn) {
            int dh = jn * 16 + l16;
            v16bf bv = ld_frag(&Vt[buf][dh * 40 + half * 8],
                               &Vt[buf][dh * 40 + 16 + half * 8]);
            accO[jn] = __builtin_amdgcn_wmma_f32_16x16x32_bf16(
                false, pa, false, bv, (short)0, accO[jn], false, false);
        }

        __syncthreads();
        buf ^= 1;
    }

    // --- finalize: multiply by 1/rowsum, store ctx (bf16) ---
#pragma unroll
    for (int jn = 0; jn < 4; ++jn) {
#pragma unroll
        for (int r = 0; r < 8; ++r) {
            float denom = rl[r];
            float v = (denom > 0.0f) ? accO[jn][r] * fast_rcp(denom) : 0.0f;
            int token = q0 + r + 8 * half;
            ctx[(size_t)(b * SS + token) * DD + h * 64 + jn * 16 + l16] = f2bf(v);
        }
    }
}

// ---------------- layernorm over rows of 1024 ----------------
__global__ __launch_bounds__(256) void layernorm_kernel(
    const float* __restrict__ in, const float* __restrict__ g,
    const float* __restrict__ bt, float* __restrict__ outF,
    unsigned short* __restrict__ outB) {
    __shared__ float red[16];
    const int tid = threadIdx.x, lane = tid & 31, wid = tid >> 5;
    const size_t row = blockIdx.x;
    const float4 v = reinterpret_cast<const float4*>(in + row * DD)[tid];
    float s  = v.x + v.y + v.z + v.w;
    float sq = v.x * v.x + v.y * v.y + v.z * v.z + v.w * v.w;
#pragma unroll
    for (int off = 16; off >= 1; off >>= 1) {
        s  += __shfl_xor(s,  off, 32);
        sq += __shfl_xor(sq, off, 32);
    }
    if (lane == 0) { red[wid] = s; red[8 + wid] = sq; }
    __syncthreads();
    float ts = 0.0f, tq = 0.0f;
#pragma unroll
    for (int w = 0; w < 8; ++w) { ts += red[w]; tq += red[8 + w]; }
    const float mean = ts * (1.0f / DD);
    const float var  = tq * (1.0f / DD) - mean * mean;
    const float rs   = rsqrtf(var + 1e-5f);
    const int c0 = tid * 4;
    float xv[4] = {v.x, v.y, v.z, v.w};
#pragma unroll
    for (int i = 0; i < 4; ++i) {
        float y = (xv[i] - mean) * rs * g[c0 + i] + bt[c0 + i];
        size_t o = row * DD + c0 + i;
        if (outF) outF[o] = y;
        if (outB) outB[o] = f2bf(y);
    }
}

// ---------------- host side ----------------
extern "C" void kernel_launch(void* const* d_in, const int* in_sizes, int n_in,
                              void* d_out, int out_size, void* d_ws, size_t ws_size,
                              hipStream_t stream) {
    (void)in_sizes; (void)n_in; (void)out_size; (void)ws_size;
    const float* x     = (const float*)d_in[0];
    const unsigned char* mask = (const unsigned char*)d_in[1];
    const float* Wq = (const float*)d_in[2];  const float* bq = (const float*)d_in[3];
    const float* Wk = (const float*)d_in[4];  const float* bk = (const float*)d_in[5];
    const float* Wv = (const float*)d_in[6];  const float* bv = (const float*)d_in[7];
    const float* Wo = (const float*)d_in[8];  const float* bo = (const float*)d_in[9];
    const float* ln1g = (const float*)d_in[10]; const float* ln1b = (const float*)d_in[11];
    const float* W1 = (const float*)d_in[12]; const float* b1 = (const float*)d_in[13];
    const float* W2 = (const float*)d_in[14]; const float* b2 = (const float*)d_in[15];
    const float* ln2g = (const float*)d_in[16]; const float* ln2b = (const float*)d_in[17];
    float* out = (float*)d_out;

    const size_t nBSD = (size_t)BS_ * DD;     // 4M
    const size_t nDD  = (size_t)DD * DD;      // 1M
    const size_t nDF  = (size_t)DD * DFFF;    // 4M
    char* p = (char*)d_ws;
    unsigned short* xb   = (unsigned short*)p; p += nBSD * 2;
    unsigned short* wqt  = (unsigned short*)p; p += nDD * 2;
    unsigned short* wkt  = (unsigned short*)p; p += nDD * 2;
    unsigned short* wvt  = (unsigned short*)p; p += nDD * 2;
    unsigned short* wot  = (unsigned short*)p; p += nDD * 2;
    unsigned short* w1t  = (unsigned short*)p; p += nDF * 2;
    unsigned short* w2t  = (unsigned short*)p; p += nDF * 2;
    unsigned short* qb   = (unsigned short*)p; p += nBSD * 2;
    unsigned short* kb_  = (unsigned short*)p; p += nBSD * 2;
    unsigned short* vb   = (unsigned short*)p; p += nBSD * 2;
    unsigned short* ctxb = (unsigned short*)p; p += nBSD * 2;
    float*          hpre = (float*)p;          p += nBSD * 4;  // reused as out_pre
    float*          hbuf = (float*)p;          p += nBSD * 4;
    unsigned short* hb   = (unsigned short*)p; p += nBSD * 2;
    unsigned short* ffb  = (unsigned short*)p; p += (size_t)BS_ * DFFF * 2;

    // casts / transposed-weight casts
    cast_bf16_kernel<<<nBSD / 256, 256, 0, stream>>>(x, xb, (int)nBSD);
    cast_transpose_kernel<<<dim3(DD / 32, DD / 32), 256, 0, stream>>>(Wq, wqt, DD, DD);
    cast_transpose_kernel<<<dim3(DD / 32, DD / 32), 256, 0, stream>>>(Wk, wkt, DD, DD);
    cast_transpose_kernel<<<dim3(DD / 32, DD / 32), 256, 0, stream>>>(Wv, wvt, DD, DD);
    cast_transpose_kernel<<<dim3(DD / 32, DD / 32), 256, 0, stream>>>(Wo, wot, DD, DD);
    cast_transpose_kernel<<<dim3(DFFF / 32, DD / 32), 256, 0, stream>>>(W1, w1t, DD, DFFF);
    cast_transpose_kernel<<<dim3(DD / 32, DFFF / 32), 256, 0, stream>>>(W2, w2t, DFFF, DD);

    // QKV projections
    dim3 gD(DD / 128, BS_ / 128);
    gemm_bf16_kernel<<<gD, 256, 0, stream>>>(xb, wqt, bq, nullptr, nullptr, qb,  BS_, DD, DD, 0);
    gemm_bf16_kernel<<<gD, 256, 0, stream>>>(xb, wkt, bk, nullptr, nullptr, kb_, BS_, DD, DD, 0);
    gemm_bf16_kernel<<<gD, 256, 0, stream>>>(xb, wvt, bv, nullptr, nullptr, vb,  BS_, DD, DD, 0);

    // flash attention
    attention_kernel<<<dim3(SS / 128, HH, BB), 256, 0, stream>>>(qb, kb_, vb, mask, ctxb);

    // output projection + residual(x), then LN1 -> h (fp32 + bf16)
    gemm_bf16_kernel<<<gD, 256, 0, stream>>>(ctxb, wot, bo, x, hpre, nullptr, BS_, DD, DD, 0);
    layernorm_kernel<<<BS_, 256, 0, stream>>>(hpre, ln1g, ln1b, hbuf, hb);

    // FFN: gelu(h W1 + b1) W2 + b2 + h, then LN2 -> out
    gemm_bf16_kernel<<<dim3(DFFF / 128, BS_ / 128), 256, 0, stream>>>(
        hb, w1t, b1, nullptr, nullptr, ffb, BS_, DFFF, DD, 1);
    gemm_bf16_kernel<<<gD, 256, 0, stream>>>(ffb, w2t, b2, hbuf, hpre, nullptr, BS_, DD, DFFF, 0);
    layernorm_kernel<<<BS_, 256, 0, stream>>>(hpre, ln2g, ln2b, out, nullptr);
}
